// GCNModelNomal_60601988546853
// MI455X (gfx1250) — matrix operand order repaired and checked by reading, not compiled, not run on hardware
//
#include <hip/hip_runtime.h>

// ---------------------------------------------------------------------------
// GCN-VAE forward for MI455X (gfx1250, wave32, WMMA, async-to-LDS).
//   h1      = relu(spmm(A, x@W1) + b1)
//   [mu|lv] = spmm(A, h1@[W2|W3]) + [b2|b3]
//   adj_rec = z @ z^T   (z = mu)  -> 1.07 GB f32 output, store-BW bound
//
// GEMMs: v_wmma_f32_16x16x32_bf16, double-buffered K-loop with
//        global_load_async_to_lds_b128 staging (ASYNCcnt pipeline).
// Weights pre-transposed to [Nc][K] so both A and B tiles stage as contiguous
// 128-bit rows and fragments load as contiguous ds_load_b128 pairs.
// SPMM: edge-parallel global_atomic_add_f32 (operands L2-resident, 192 MB L2).
// ---------------------------------------------------------------------------

typedef __bf16 v16bf __attribute__((ext_vector_type(16)));
typedef float  v8f   __attribute__((ext_vector_type(8)));

union FragBF { unsigned u[8]; v16bf v; };

constexpr int Nn  = 16384;
constexpr int Ee  = 524288;
constexpr int FIN = 512;
constexpr int Hh  = 256;
constexpr int Oo  = 64;

__device__ __forceinline__ unsigned short f2bf(float f) {
  unsigned u = __float_as_uint(f);
  u += 0x7FFFu + ((u >> 16) & 1u);   // round-to-nearest-even
  return (unsigned short)(u >> 16);
}

// --- CDNA5 async global->LDS copy helpers (ASYNCcnt-tracked) ----------------
typedef __attribute__((address_space(3))) unsigned short lds_ushort;

__device__ __forceinline__ unsigned lds_off_of(unsigned short* p) {
  // generic->local addrspacecast truncates to the 32-bit LDS byte offset
  return (unsigned)(unsigned long long)(lds_ushort*)p;
}

__device__ __forceinline__ void async_copy_b128(unsigned short* lds_dst,
                                                const unsigned short* gsrc) {
  unsigned loff = lds_off_of(lds_dst);
  asm volatile("global_load_async_to_lds_b128 %0, %1, off"
               :: "v"(loff), "v"(gsrc) : "memory");
}

__device__ __forceinline__ void wait_async0() {
  asm volatile("s_wait_asynccnt 0x0" ::: "memory");
}

// ---------------------------------------------------------------- converters
__global__ void cvt_bf16_kernel(const float* __restrict__ src,
                                unsigned short* __restrict__ dst, int n) {
  int i = blockIdx.x * blockDim.x + threadIdx.x;
  int stride = gridDim.x * blockDim.x;
  for (; i < n; i += stride) dst[i] = f2bf(src[i]);
}

// W1t[n*FIN + k] = bf16(W1[k*Hh + n])  : [256][512]
__global__ void cvt_w1t_kernel(const float* __restrict__ W1,
                               unsigned short* __restrict__ Wt) {
  int i = blockIdx.x * blockDim.x + threadIdx.x;   // over Hh*FIN
  if (i >= Hh * FIN) return;
  int n = i / FIN, k = i - n * FIN;
  Wt[i] = f2bf(W1[(size_t)k * Hh + n]);
}

// W23t[n*Hh + k] = bf16( n<64 ? W2[k,n] : W3[k,n-64] )  : [128][256]
__global__ void pack_w23t_kernel(const float* __restrict__ W2,
                                 const float* __restrict__ W3,
                                 unsigned short* __restrict__ Wt) {
  int i = blockIdx.x * blockDim.x + threadIdx.x;   // over 128*Hh
  if (i >= 128 * Hh) return;
  int n = i >> 8, k = i & 255;
  float v = (n < Oo) ? W2[(size_t)k * Oo + n] : W3[(size_t)k * Oo + (n - Oo)];
  Wt[i] = f2bf(v);
}

__global__ void zero_kernel(float* __restrict__ p, int n) {
  int i = blockIdx.x * blockDim.x + threadIdx.x;
  int stride = gridDim.x * blockDim.x;
  for (; i < n; i += stride) p[i] = 0.0f;
}

// ------------------------------------------------------------ bf16 WMMA GEMM
// C[M,Nc] = A[M,K] @ Bt[Nc,K]^T, bf16 in / f32 out.  Block = 128x128 tile,
// 8 waves, each wave = 64x32 slab = 4x2 grid of 16x16 WMMA tiles, BK = 32.
// Double-buffered LDS, staged with global_load_async_to_lds_b128.
__launch_bounds__(256)
__global__ void gemm_bf16_kernel(const unsigned short* __restrict__ A,
                                 const unsigned short* __restrict__ Bt,
                                 float* __restrict__ C, int K, int Nc) {
  __shared__ unsigned short As[2][128 * 32];   // [row][k]
  __shared__ unsigned short Bs[2][128 * 32];   // [col][k]

  const int tid  = threadIdx.x;
  const int lane = tid & 31;
  const int wave = tid >> 5;
  const int wm   = wave & 1;      // 2 M-slabs of 64 rows
  const int wn   = wave >> 1;     // 4 N-slabs of 32 cols
  const int m0   = blockIdx.x * 128;
  const int n0   = blockIdx.y * 128;
  const int half = lane >> 4;     // 0/1 lane half
  const int r15  = lane & 15;

  v8f acc[4][2] = {};

  const int ar = (tid + 0)   >> 2, as = tid & 3;          // chunk 0
  const int br = (tid + 256) >> 2;                        // chunk 1 (same &3)

  // issue one K-slice's worth of async copies into buffer `buf`
  auto stage = [&](int buf, int k0) {
    async_copy_b128(&As[buf][ar * 32 + as * 8],
                    A  + (size_t)(m0 + ar) * K + k0 + as * 8);
    async_copy_b128(&Bs[buf][ar * 32 + as * 8],
                    Bt + (size_t)(n0 + ar) * K + k0 + as * 8);
    async_copy_b128(&As[buf][br * 32 + as * 8],
                    A  + (size_t)(m0 + br) * K + k0 + as * 8);
    async_copy_b128(&Bs[buf][br * 32 + as * 8],
                    Bt + (size_t)(n0 + br) * K + k0 + as * 8);
  };

  const int nk = K >> 5;
  stage(0, 0);

  for (int t = 0; t < nk; ++t) {
    const int buf = t & 1;
    wait_async0();          // own async copies have landed in LDS
    __syncthreads();        // everyone's copies have landed
    if (t + 1 < nk) stage(buf ^ 1, (t + 1) * 32);
    if (t + 2 < nk) {       // warm K-slice beyond the double buffer
      __builtin_prefetch(A  + (size_t)(m0 + (tid >> 1)) * K + (t + 2) * 32, 0, 0);
      __builtin_prefetch(Bt + (size_t)(n0 + (tid >> 1)) * K + (t + 2) * 32, 0, 0);
    }

    // ---- A fragments: contiguous k = 16*half + s (matches B) ----
    FragBF afrag[4];
#pragma unroll
    for (int mt = 0; mt < 4; ++mt) {
      const unsigned short* arow = &As[buf][(wm * 64 + mt * 16 + r15) * 32];
#pragma unroll
      for (int v = 0; v < 8; ++v) {
        int kk = half * 16 + v * 2;
        afrag[mt].u[v] = *(const unsigned*)&arow[kk];
      }
    }
    // ---- B fragments: [col][k] layout -> contiguous 32B per fragment ----
    FragBF bfrag[2];
#pragma unroll
    for (int nt = 0; nt < 2; ++nt) {
      const unsigned short* brow = &Bs[buf][(wn * 32 + nt * 16 + r15) * 32];
#pragma unroll
      for (int v = 0; v < 8; ++v)
        bfrag[nt].u[v] = *(const unsigned*)&brow[half * 16 + v * 2];
    }
#pragma unroll
    for (int mt = 0; mt < 4; ++mt)
#pragma unroll
      for (int nt = 0; nt < 2; ++nt)
        acc[mt][nt] = __builtin_amdgcn_wmma_f32_16x16x32_bf16(
            false, afrag[mt].v, false, bfrag[nt].v,
            (short)0, acc[mt][nt], false, false);
  }

  // ---- store D: VGPR v -> row v + 8*half, col = lane&15 ----
#pragma unroll
  for (int mt = 0; mt < 4; ++mt)
#pragma unroll
    for (int nt = 0; nt < 2; ++nt)
#pragma unroll
      for (int v = 0; v < 8; ++v) {
        int row = m0 + wm * 64 + mt * 16 + v + 8 * half;
        int col = n0 + wn * 32 + nt * 16 + r15;
        C[(size_t)row * Nc + col] = acc[mt][nt][v];
      }
}

// ------------------------------------------------------------------- SPMM
// One wave per edge: out[row, :] += val * dense[col, :] via f32 atomics.
// dense (<=16.8MB) and accumulator live in the 192MB L2.
__global__ void spmm_kernel(const int* __restrict__ rows,
                            const int* __restrict__ cols,
                            const float* __restrict__ vals,
                            const float* __restrict__ dense,
                            float* __restrict__ out, int D, int E) {
  int e = (int)((blockIdx.x * (size_t)blockDim.x + threadIdx.x) >> 5);
  int lane = threadIdx.x & 31;
  if (e >= E) return;
  int r = rows[e], c = cols[e];
  float v = vals[e];
  const float* src = dense + (size_t)c * D;
  float* dst = out + (size_t)r * D;
  for (int j = lane; j < D; j += 32) atomicAdd(&dst[j], v * src[j]);
}

// ------------------------------------------------- bias + relu + bf16 cast
__global__ void bias_relu_cvt_kernel(const float* __restrict__ acc,
                                     const float* __restrict__ b1,
                                     unsigned short* __restrict__ h1b, int n) {
  int i = blockIdx.x * blockDim.x + threadIdx.x;
  int stride = gridDim.x * blockDim.x;
  for (; i < n; i += stride) {
    float h = acc[i] + b1[i & (Hh - 1)];
    h1b[i] = f2bf(fmaxf(h, 0.0f));
  }
}

// ---------------------------- split [mu|logvar], emit mu/logvar/z + bf16 z
__global__ void finalize_kernel(const float* __restrict__ acc,
                                const float* __restrict__ b2,
                                const float* __restrict__ b3,
                                float* __restrict__ mu,
                                float* __restrict__ logvar,
                                float* __restrict__ z,
                                unsigned short* __restrict__ zb) {
  int i = blockIdx.x * blockDim.x + threadIdx.x;   // over N*128
  if (i >= Nn * 128) return;
  int n = i >> 7, j = i & 127;
  float v = acc[i];
  if (j < Oo) {
    float m = v + b2[j];
    mu[(size_t)n * Oo + j] = m;
    z[(size_t)n * Oo + j]  = m;
    zb[(size_t)n * Oo + j] = f2bf(m);
  } else {
    logvar[(size_t)n * Oo + (j - Oo)] = v + b3[j - Oo];
  }
}

// --------------------------------------------------------- decoder z @ z^T
// 128x128 output tile per block; K=64 fully in LDS (async-staged);
// 16 WMMAs per wave; bound by 1.07GB of f32 stores (~46us at 23.3 TB/s).
__launch_bounds__(256)
__global__ void decoder_kernel(const unsigned short* __restrict__ Z,
                               float* __restrict__ C) {
  __shared__ unsigned short Azs[128 * 64];
  __shared__ unsigned short Bzs[128 * 64];

  const int tid  = threadIdx.x;
  const int lane = tid & 31;
  const int wave = tid >> 5;
  const int wm   = wave & 1;
  const int wn   = wave >> 1;
  const int m0   = blockIdx.x * 128;
  const int n0   = blockIdx.y * 128;
  const int half = lane >> 4;
  const int r15  = lane & 15;

#pragma unroll
  for (int i = 0; i < 4; ++i) {
    int idx = tid + i * 256;                 // 0..1023 16B slots per tile
    int rr = idx >> 3, ss = idx & 7;
    async_copy_b128(&Azs[rr * 64 + ss * 8],
                    Z + (size_t)(m0 + rr) * 64 + ss * 8);
    async_copy_b128(&Bzs[rr * 64 + ss * 8],
                    Z + (size_t)(n0 + rr) * 64 + ss * 8);
  }
  wait_async0();
  __syncthreads();

  v8f acc[4][2] = {};
#pragma unroll
  for (int kk = 0; kk < 64; kk += 32) {
    FragBF afrag[4];
#pragma unroll
    for (int mt = 0; mt < 4; ++mt) {
      const unsigned short* arow = &Azs[(wm * 64 + mt * 16 + r15) * 64 + kk];
#pragma unroll
      for (int v = 0; v < 8; ++v) {
        int ko = half * 16 + v * 2;
        afrag[mt].u[v] = *(const unsigned*)&arow[ko];
      }
    }
    FragBF bfrag[2];
#pragma unroll
    for (int nt = 0; nt < 2; ++nt) {
      const unsigned short* brow = &Bzs[(wn * 32 + nt * 16 + r15) * 64 + kk];
#pragma unroll
      for (int v = 0; v < 8; ++v)
        bfrag[nt].u[v] = *(const unsigned*)&brow[half * 16 + v * 2];
    }
#pragma unroll
    for (int mt = 0; mt < 4; ++mt)
#pragma unroll
      for (int nt = 0; nt < 2; ++nt)
        acc[mt][nt] = __builtin_amdgcn_wmma_f32_16x16x32_bf16(
            false, afrag[mt].v, false, bfrag[nt].v,
            (short)0, acc[mt][nt], false, false);
  }

#pragma unroll
  for (int mt = 0; mt < 4; ++mt)
#pragma unroll
    for (int nt = 0; nt < 2; ++nt)
#pragma unroll
      for (int v = 0; v < 8; ++v) {
        int row = m0 + wm * 64 + mt * 16 + v + 8 * half;
        int col = n0 + wn * 32 + nt * 16 + r15;
        C[(size_t)row * Nn + col] = acc[mt][nt][v];
      }
}

// ------------------------------------------------------------------ launch
extern "C" void kernel_launch(void* const* d_in, const int* in_sizes, int n_in,
                              void* d_out, int out_size, void* d_ws,
                              size_t ws_size, hipStream_t stream) {
  const float* x    = (const float*)d_in[0];
  const int*   rows = (const int*)d_in[1];
  const int*   cols = (const int*)d_in[2];
  const float* vals = (const float*)d_in[3];
  const float* W1   = (const float*)d_in[4];
  const float* b1   = (const float*)d_in[5];
  const float* W2   = (const float*)d_in[6];
  const float* b2   = (const float*)d_in[7];
  const float* W3   = (const float*)d_in[8];
  const float* b3   = (const float*)d_in[9];

  char* ws = (char*)d_ws;
  size_t off = 0;
  auto alloc = [&](size_t bytes) -> char* {
    char* p = ws + off;
    off = (off + bytes + 255) & ~(size_t)255;
    return p;
  };

  unsigned short* xb   = (unsigned short*)alloc((size_t)Nn * FIN * 2);
  unsigned short* W1t  = (unsigned short*)alloc((size_t)Hh * FIN * 2);
  unsigned short* W23t = (unsigned short*)alloc((size_t)128 * Hh * 2);
  float*          xw1  = (float*)alloc((size_t)Nn * Hh * 4);   // reused as t2
  float*          acc1 = (float*)alloc((size_t)Nn * Hh * 4);   // reused as acc2
  unsigned short* h1b  = (unsigned short*)alloc((size_t)Nn * Hh * 2);
  unsigned short* zb   = (unsigned short*)alloc((size_t)Nn * Oo * 2);
  float* t2   = xw1;    // [N,128], alias: xw1 dead after first SPMM
  float* acc2 = acc1;   // [N,128], alias: acc1 dead after bias/relu

  float* adj_rec = (float*)d_out;
  float* mu      = adj_rec + (size_t)Nn * Nn;
  float* logvar  = mu + (size_t)Nn * Oo;
  float* zout    = logvar + (size_t)Nn * Oo;

  // 1) bf16 casts of GEMM operands (weights transposed to [Nc][K])
  cvt_bf16_kernel<<<4096, 256, 0, stream>>>(x, xb, Nn * FIN);
  cvt_w1t_kernel<<<(Hh * FIN) / 256, 256, 0, stream>>>(W1, W1t);
  pack_w23t_kernel<<<(128 * Hh) / 256, 256, 0, stream>>>(W2, W3, W23t);

  // 2) xw1 = x @ W1   (16384x512x256, bf16 WMMA, async-staged)
  gemm_bf16_kernel<<<dim3(Nn / 128, Hh / 128), 256, 0, stream>>>(
      xb, W1t, xw1, FIN, Hh);

  // 3) acc1 = spmm(A, xw1)
  zero_kernel<<<4096, 256, 0, stream>>>(acc1, Nn * Hh);
  spmm_kernel<<<Ee / 8, 256, 0, stream>>>(rows, cols, vals, xw1, acc1, Hh, Ee);

  // 4) h1b = bf16(relu(acc1 + b1))
  bias_relu_cvt_kernel<<<4096, 256, 0, stream>>>(acc1, b1, h1b, Nn * Hh);

  // 5) t2 = h1 @ [W2|W3]   (16384x256x128)
  gemm_bf16_kernel<<<dim3(Nn / 128, 1), 256, 0, stream>>>(h1b, W23t, t2, Hh, 128);

  // 6) acc2 = spmm(A, t2)
  zero_kernel<<<2048, 256, 0, stream>>>(acc2, Nn * 128);
  spmm_kernel<<<Ee / 8, 256, 0, stream>>>(rows, cols, vals, t2, acc2, 128, Ee);

  // 7) mu / logvar / z outputs + bf16 z
  finalize_kernel<<<(Nn * 128) / 256, 256, 0, stream>>>(
      acc2, b2, b3, mu, logvar, zout, zb);

  // 8) adj_rec = z @ z^T  (store-bandwidth bound)
  decoder_kernel<<<dim3(Nn / 128, Nn / 128), 256, 0, stream>>>(zb, adj_rec);
}
